// _HAN_25211458027618
// MI455X (gfx1250) — matrix operand order repaired and checked by reading, not compiled, not run on hardware
//
#include <hip/hip_runtime.h>
#include <hip/hip_bf16.h>

// ---------------- problem constants (from reference) ----------------
static constexpr int Nn   = 8192;     // nodes
static constexpr int IND  = 512;      // input feature dim
static constexpr int Hh   = 8;        // heads
static constexpr int HID  = 64;       // per-head dim
static constexpr int Dd   = Hh * HID; // 512
static constexpr int NE   = 262144;   // edges (symmetric)
static constexpr int NEH  = 131072;   // half edges; reverse edge = e ^ NEH
static constexpr int Tk   = 10;       // top-K threshold
static constexpr float NEGV = -9.0e15f;

typedef _Float16 half16 __attribute__((ext_vector_type(16)));
typedef float    float8 __attribute__((ext_vector_type(8)));

// ---------------- tiny utility kernels ----------------
__global__ void fill_u32_kernel(unsigned* p, unsigned v, int n) {
  int i = blockIdx.x * blockDim.x + threadIdx.x;
  if (i < n) p[i] = v;
}

__global__ void f32_to_f16_kernel(const float* in, _Float16* out, int n) {
  int i = blockIdx.x * blockDim.x + threadIdx.x;
  if (i < n) out[i] = (_Float16)in[i];
}

// ---------------- WMMA GEMM: C(MxNcol) = A(MxK) * B(KxNcol), f16 in / f32 out ----
// Block = 256 thr = 8 waves; block tile 128x64; wave tile 32x32 = 2x2 WMMA 16x16x32.
// A tile staged row-major (fragment rows are contiguous -> ds_load_b128).
// B tile staged TRANSPOSED (BsT[col][k]) so each lane's B fragment (two runs of 8 consecutive
// K values at a fixed column) is two 16-byte runs -> 2x ds_load_b128 instead of
// 16x ds_load_u16 + v_mov_b16 packing in the WMMA dependency chain.
__global__ __launch_bounds__(256) void wmma_gemm_kernel(
    const _Float16* __restrict__ A, const _Float16* __restrict__ B,
    float* __restrict__ C, int M, int K, int Ncol) {
  __shared__ _Float16 As [128][40];  // [row][k]  pad->16B-aligned rows
  __shared__ _Float16 BsT[ 64][40];  // [col][k]  transposed B tile, same padding

  const int tid  = threadIdx.x;
  const int lane = tid & 31;
  const int wave = tid >> 5;
  const int wm   = (wave >> 1) * 32;   // wave row inside block tile (0..96)
  const int wn   = (wave & 1) * 32;    // wave col inside block tile (0 or 32)
  const int bM   = blockIdx.x * 128;
  const int bN   = blockIdx.y * 64;

  float8 acc[2][2] = {};               // zero-init accumulators (8 VGPR each)

  for (int k0 = 0; k0 < K; k0 += 32) {
    // ---- stage A tile 128x32 (4096 halves, 16/thread, 8-half contiguous runs) ----
#pragma unroll
    for (int r = 0; r < 2; ++r) {
      int idx = tid * 8 + r * 2048;
      int row = idx >> 5, col = idx & 31;
      const _Float16* gp = A + (size_t)(bM + row) * K + k0 + col;
#pragma unroll
      for (int j = 0; j < 8; ++j) As[row][col + j] = gp[j];
    }
    // ---- stage B tile 32x64, transposed into LDS ----
    // thread loads 8 contiguous halves of one B row (coalesced global_load_b128),
    // scatters them across 8 BsT columns (8x ds_store_b16).
    {
      int idx = tid * 8;
      int row = idx >> 6, col = idx & 63;  // row = k (0..31), col = n (0..56 step 8)
      const _Float16* gp = B + (size_t)(k0 + row) * Ncol + bN + col;
#pragma unroll
      for (int j = 0; j < 8; ++j) BsT[col + j][row] = gp[j];
    }
    if (k0 + 32 < K)  // lowers to global_prefetch_b8 on gfx1250
      __builtin_prefetch(A + (size_t)(bM + (tid >> 2)) * K + k0 + 32, 0, 0);
    __syncthreads();

    // ---- gather fragments per documented 16-bit layouts ----
    half16 af[2], bf[2];
#pragma unroll
    for (int mi = 0; mi < 2; ++mi) {
      // A 16x32: lane&15 = M row; lanes<16 -> K=0..7 / 16..23, lanes>=16 -> K=8..15 / 24..31
      int mrow = wm + mi * 16 + (lane & 15);
      int kb   = (lane < 16) ? 0 : 8;
      const _Float16* ap = &As[mrow][0];
#pragma unroll
      for (int j = 0; j < 8; ++j) af[mi][j]     = ap[kb + j];
#pragma unroll
      for (int j = 0; j < 8; ++j) af[mi][8 + j] = ap[kb + 16 + j];
    }
#pragma unroll
    for (int ni = 0; ni < 2; ++ni) {
      // B 32x16: lane&15 = N col; lanes<16 -> K=0..7 / 16..23, lanes>=16 -> K=8..15 / 24..31
      int ncol = wn + ni * 16 + (lane & 15);
      int kb   = (lane < 16) ? 0 : 8;
      const _Float16* bp = &BsT[ncol][kb];  // two 8-half runs (16B each, 16B-aligned)
#pragma unroll
      for (int j = 0; j < 16; ++j) bf[ni][j] = bp[j < 8 ? j : j + 8];
    }
#pragma unroll
    for (int mi = 0; mi < 2; ++mi)
#pragma unroll
      for (int ni = 0; ni < 2; ++ni)
        acc[mi][ni] = __builtin_amdgcn_wmma_f32_16x16x32_f16(
            false, af[mi], false, bf[ni], (short)0, acc[mi][ni], false, false);
    __syncthreads();
  }

  // ---- store f32 C per documented layout: lane&15 = N, VGPR v -> M=v (+8 upper lanes)
#pragma unroll
  for (int mi = 0; mi < 2; ++mi)
#pragma unroll
    for (int ni = 0; ni < 2; ++ni) {
      int n     = bN + wn + ni * 16 + (lane & 15);
      int mbase = bM + wm + mi * 16 + ((lane < 16) ? 0 : 8);
#pragma unroll
      for (int v = 0; v < 8; ++v)
        C[(size_t)(mbase + v) * Ncol + n] = acc[mi][ni][v];
    }
}

// ---------------- per-edge per-head dot products: one wave32 per edge ----------
__global__ __launch_bounds__(256) void edge_dot_kernel(
    const int* __restrict__ src, const int* __restrict__ dst,
    const float* __restrict__ f, float* __restrict__ eArr) {
  int w    = (blockIdx.x * blockDim.x + threadIdx.x) >> 5;  // edge id
  int lane = threadIdx.x & 31;
  const float* fa = f + (size_t)src[w] * Dd + lane * 16;
  const float* fb = f + (size_t)dst[w] * Dd + lane * 16;
  float p = 0.f;
#pragma unroll
  for (int j = 0; j < 16; ++j) p += fa[j] * fb[j];
  // lane covers 16 dims = quarter of one head; reduce groups of 4 lanes
  p += __shfl_xor(p, 1, 32);
  p += __shfl_xor(p, 2, 32);
  if ((lane & 3) == 0) eArr[(size_t)w * Hh + (lane >> 2)] = p;
}

__global__ void edge_vals_kernel(const float* __restrict__ trans,
                                 const float* __restrict__ eArr,
                                 float* __restrict__ vals) {
  int e = blockIdx.x * blockDim.x + threadIdx.x;
  float s = 0.f;
#pragma unroll
  for (int h = 0; h < Hh; ++h) s += eArr[(size_t)e * Hh + h];
  vals[e] = trans[e] * s;
}

// ---------------- CSR build: count / scan / scatter ----------------
__global__ void count_kernel(const int* __restrict__ src, const int* __restrict__ dst,
                             int* cntS, int* cntD) {
  int e = blockIdx.x * blockDim.x + threadIdx.x;
  atomicAdd(&cntS[src[e]], 1);
  atomicAdd(&cntD[dst[e]], 1);
}

__global__ __launch_bounds__(1024) void scan_kernel(const int* __restrict__ cntS,
                                                    const int* __restrict__ cntD,
                                                    int* offS, int* offD) {
  const int* cnt = blockIdx.x ? cntD : cntS;
  int*       off = blockIdx.x ? offD : offS;
  __shared__ int part[1024];
  int t = threadIdx.x;
  int v[8]; int tot = 0;
#pragma unroll
  for (int i = 0; i < 8; ++i) { v[i] = cnt[t * 8 + i]; tot += v[i]; }
  part[t] = tot;
  __syncthreads();
  for (int d = 1; d < 1024; d <<= 1) {
    int add = (t >= d) ? part[t - d] : 0;
    __syncthreads();
    part[t] += add;
    __syncthreads();
  }
  int run = part[t] - tot;  // exclusive base
#pragma unroll
  for (int i = 0; i < 8; ++i) { off[t * 8 + i] = run; run += v[i]; }
  if (t == 1023) off[Nn] = run;
}

__global__ void scatter_kernel(const int* __restrict__ src, const int* __restrict__ dst,
                               const int* __restrict__ offS, const int* __restrict__ offD,
                               int* curS, int* curD, int* listS, int* listD) {
  int e = blockIdx.x * blockDim.x + threadIdx.x;
  int s = src[e], d = dst[e];
  int ps = atomicAdd(&curS[s], 1);
  listS[offS[s] + ps] = e;
  int pd = atomicAdd(&curD[d], 1);
  listD[offD[d] + pd] = e;
}

// ---------------- per-row (by src) T-th-largest threshold: one wave32 per row ----
// A row has ~32 nonzeros out of 8192 -> T-th largest incl. zeros is the T-th largest
// POSITIVE nonzero if >=T positives exist, else 0. Never touch the dense matrix.
__global__ __launch_bounds__(256) void row_thr_kernel(
    const int* __restrict__ offS, const int* __restrict__ listS,
    const float* __restrict__ vals, float* __restrict__ thr) {
  int row  = (blockIdx.x * blockDim.x + threadIdx.x) >> 5;
  int lane = threadIdx.x & 31;
  int beg = offS[row], cnt = offS[row + 1] - beg;

  float v[16]; int nl = 0;
  for (int i = lane; i < cnt && nl < 16; i += 32) v[nl++] = vals[listS[beg + i]];
  int pos = 0;
  for (int i = 0; i < nl; ++i) if (v[i] > 0.f) pos++;
  for (int o = 16; o; o >>= 1) pos += __shfl_xor(pos, o, 32);

  float kth = 0.f; unsigned marked = 0; int found = 0;
  for (int t = 0; t < Tk; ++t) {
    float bv = -3.0e38f; int bi = -1;
    for (int i = 0; i < nl; ++i)
      if (!((marked >> i) & 1) && v[i] > bv) { bv = v[i]; bi = i; }
    float rv = bv; int rl = lane;
    for (int o = 16; o; o >>= 1) {
      float ov = __shfl_xor(rv, o, 32);
      int   ol = __shfl_xor(rl, o, 32);
      if (ov > rv || (ov == rv && ol < rl)) { rv = ov; rl = ol; }
    }
    if (rv <= -3.0e38f) break;
    if (rl == lane && bi >= 0) marked |= (1u << bi);
    kth = rv; ++found;
  }
  if (lane == 0) thr[row] = (pos >= Tk && found >= Tk) ? fmaxf(kth, 0.f) : 0.f;
}

// ---------------- per-dst-row softmax + aggregation + ELU: one wave32 per row ----
__global__ __launch_bounds__(256) void row_agg_kernel(
    const int* __restrict__ offD, const int* __restrict__ listD,
    const int* __restrict__ src, const float* __restrict__ eArr,
    const float* __restrict__ vals, const float* __restrict__ thr,
    const float* __restrict__ f, float* __restrict__ z) {
  int row  = (blockIdx.x * blockDim.x + threadIdx.x) >> 5;
  int lane = threadIdx.x & 31;
  int beg = offD[row], cnt = offD[row + 1] - beg;
  float th = thr[row];  // gathered = A[dst,src] ~= vals[reverse edge]; mask vs thr[dst]

  // phase 1: per-head max (edge-softmax stabilization)
  float m[Hh];
#pragma unroll
  for (int h = 0; h < Hh; ++h) m[h] = -3.0e38f;
  for (int i = lane; i < cnt; i += 32) {
    int e = listD[beg + i];
    bool msk = vals[e ^ NEH] < th;
    const float* ep = eArr + (size_t)e * Hh;
#pragma unroll
    for (int h = 0; h < Hh; ++h) m[h] = fmaxf(m[h], msk ? NEGV : ep[h]);
  }
  for (int o = 16; o; o >>= 1)
#pragma unroll
    for (int h = 0; h < Hh; ++h) m[h] = fmaxf(m[h], __shfl_xor(m[h], o, 32));

  // phase 2: per-head exp-sums
  float s[Hh];
#pragma unroll
  for (int h = 0; h < Hh; ++h) s[h] = 0.f;
  for (int i = lane; i < cnt; i += 32) {
    int e = listD[beg + i];
    bool msk = vals[e ^ NEH] < th;
    const float* ep = eArr + (size_t)e * Hh;
#pragma unroll
    for (int h = 0; h < Hh; ++h) s[h] += __expf((msk ? NEGV : ep[h]) - m[h]);
  }
  for (int o = 16; o; o >>= 1)
#pragma unroll
    for (int h = 0; h < Hh; ++h) s[h] += __shfl_xor(s[h], o, 32);

  // phase 3: aggregate f[src]*a, lane owns 16 contiguous dims (single head each)
  int h = lane >> 2;
  float inv = (s[h] > 0.f) ? 1.f / s[h] : 0.f;
  float acc[16];
#pragma unroll
  for (int j = 0; j < 16; ++j) acc[j] = 0.f;
  for (int i = 0; i < cnt; ++i) {
    int e = listD[beg + i];
    bool msk = vals[e ^ NEH] < th;
    float ev  = msk ? NEGV : eArr[(size_t)e * Hh + h];
    float att = __expf(ev - m[h]) * inv;
    const float* fs = f + (size_t)src[e] * Dd + lane * 16;
#pragma unroll
    for (int j = 0; j < 16; ++j) acc[j] += fs[j] * att;
  }
  float* zr = z + (size_t)row * Dd + lane * 16;
#pragma unroll
  for (int j = 0; j < 16; ++j) {
    float x = acc[j];
    zr[j] = (x > 0.f) ? x : (__expf(x) - 1.f);  // ELU
  }
}

// ---------------- semantic attention score accumulation: wave per (node, metapath) --
__global__ __launch_bounds__(256) void sem_kernel(
    const float* __restrict__ z0, const float* __restrict__ z1,
    const float* __restrict__ w1, const float* __restrict__ b1,
    const float* __restrict__ w2, float* __restrict__ wmean) {
  int gw   = (blockIdx.x * blockDim.x + threadIdx.x) >> 5;
  int lane = threadIdx.x & 31;
  int node = gw >> 1, mp = gw & 1;
  const float* zr = (mp ? z1 : z0) + (size_t)node * Dd;
  float acc[4] = {0.f, 0.f, 0.f, 0.f};
  for (int k = 0; k < Dd; ++k) {
    float zk = zr[k];
    const float* wr = w1 + k * 128;
#pragma unroll
    for (int t = 0; t < 4; ++t) acc[t] += zk * wr[lane + 32 * t];
  }
  float wsum = 0.f;
#pragma unroll
  for (int t = 0; t < 4; ++t) {
    int j = lane + 32 * t;
    wsum += tanhf(acc[t] + b1[j]) * w2[j];
  }
  for (int o = 16; o; o >>= 1) wsum += __shfl_xor(wsum, o, 32);
  if (lane == 0) atomicAdd(&wmean[mp], wsum);
}

// ---------------- beta softmax + weighted sum + final linear: wave per node --------
__global__ __launch_bounds__(256) void final_kernel(
    const float* __restrict__ z0, const float* __restrict__ z1,
    const float* __restrict__ wmean, const float* __restrict__ lw,
    const float* __restrict__ lb, float* __restrict__ out) {
  int node = (blockIdx.x * blockDim.x + threadIdx.x) >> 5;
  int lane = threadIdx.x & 31;
  float w0 = wmean[0] * (1.f / (float)Nn), w1 = wmean[1] * (1.f / (float)Nn);
  float mx = fmaxf(w0, w1);
  float e0 = __expf(w0 - mx), e1 = __expf(w1 - mx);
  float b0 = e0 / (e0 + e1), b1 = e1 / (e0 + e1);
  float c0 = 0.f, c1 = 0.f, c2 = 0.f;
  for (int d = lane; d < Dd; d += 32) {
    float hv = b0 * z0[(size_t)node * Dd + d] + b1 * z1[(size_t)node * Dd + d];
    c0 += hv * lw[d * 3 + 0];
    c1 += hv * lw[d * 3 + 1];
    c2 += hv * lw[d * 3 + 2];
  }
  for (int o = 16; o; o >>= 1) {
    c0 += __shfl_xor(c0, o, 32);
    c1 += __shfl_xor(c1, o, 32);
    c2 += __shfl_xor(c2, o, 32);
  }
  if (lane == 0) {
    out[node * 3 + 0] = c0 + lb[0];
    out[node * 3 + 1] = c1 + lb[1];
    out[node * 3 + 2] = c2 + lb[2];
  }
}

// ---------------- orchestration ----------------
extern "C" void kernel_launch(void* const* d_in, const int* in_sizes, int n_in,
                              void* d_out, int out_size, void* d_ws, size_t ws_size,
                              hipStream_t stream) {
  (void)in_sizes; (void)n_in; (void)out_size; (void)ws_size;
  const float* feat     = (const float*)d_in[0];
  const int*   srcA[2]  = {(const int*)d_in[1], (const int*)d_in[4]};
  const int*   dstA[2]  = {(const int*)d_in[2], (const int*)d_in[5]};
  const float* transA[2]= {(const float*)d_in[3], (const float*)d_in[6]};
  const float* fcwA[2]  = {(const float*)d_in[7], (const float*)d_in[8]};
  const float* semW1    = (const float*)d_in[9];
  const float* semB1    = (const float*)d_in[10];
  const float* semW2    = (const float*)d_in[11];
  const float* linW     = (const float*)d_in[12];
  const float* linB     = (const float*)d_in[13];
  float* out = (float*)d_out;

  char* ws = (char*)d_ws;
  size_t o = 0;
  auto take = [&](size_t bytes) -> char* {
    char* r = ws + o;
    o += (bytes + 255) & ~(size_t)255;
    return r;
  };
  _Float16* featH = (_Float16*)take((size_t)Nn * IND * 2);
  _Float16* wH    = (_Float16*)take((size_t)IND * Dd * 2);
  float*    fbuf  = (float*)take((size_t)Nn * Dd * 4);     // projected feats (per conv)
  float*    eArr  = (float*)take((size_t)NE * Hh * 4);
  float*    vals  = (float*)take((size_t)NE * 4);
  int*      cnt4  = (int*)take((size_t)4 * Nn * 4);        // cntS,cntD,curS,curD contiguous
  int*      cntS = cnt4, *cntD = cnt4 + Nn, *curS = cnt4 + 2 * Nn, *curD = cnt4 + 3 * Nn;
  int*      offS  = (int*)take((size_t)(Nn + 1) * 4);
  int*      offD  = (int*)take((size_t)(Nn + 1) * 4);
  int*      listS = (int*)take((size_t)NE * 4);
  int*      listD = (int*)take((size_t)NE * 4);
  float*    thr   = (float*)take((size_t)Nn * 4);
  float*    z0    = (float*)take((size_t)Nn * Dd * 4);
  float*    z1    = (float*)take((size_t)Nn * Dd * 4);
  float*    wmean = (float*)take(256);

  // feat -> f16 once
  f32_to_f16_kernel<<<(Nn * IND) / 256, 256, 0, stream>>>(feat, featH, Nn * IND);

  for (int c = 0; c < 2; ++c) {
    float* zc = c ? z1 : z0;
    // 1. weights -> f16, WMMA GEMM projection f = feat @ fc_w
    f32_to_f16_kernel<<<(IND * Dd) / 256, 256, 0, stream>>>(fcwA[c], wH, IND * Dd);
    wmma_gemm_kernel<<<dim3(Nn / 128, Dd / 64, 1), 256, 0, stream>>>(
        featH, wH, fbuf, Nn, IND, Dd);
    // 2. per-edge per-head dots, edge values
    edge_dot_kernel<<<NE / 8, 256, 0, stream>>>(srcA[c], dstA[c], fbuf, eArr);
    edge_vals_kernel<<<NE / 256, 256, 0, stream>>>(transA[c], eArr, vals);
    // 3. CSR by src (for top-T threshold) and by dst (for softmax/agg)
    fill_u32_kernel<<<(4 * Nn) / 256, 256, 0, stream>>>((unsigned*)cnt4, 0u, 4 * Nn);
    count_kernel<<<NE / 256, 256, 0, stream>>>(srcA[c], dstA[c], cntS, cntD);
    scan_kernel<<<2, 1024, 0, stream>>>(cntS, cntD, offS, offD);
    scatter_kernel<<<NE / 256, 256, 0, stream>>>(srcA[c], dstA[c], offS, offD,
                                                 curS, curD, listS, listD);
    // 4. per-row threshold (sparse top-T; dense 256MB A never materialized)
    row_thr_kernel<<<Nn / 8, 256, 0, stream>>>(offS, listS, vals, thr);
    // 5. per-dst-row masked edge-softmax + aggregation + ELU
    row_agg_kernel<<<Nn / 8, 256, 0, stream>>>(offD, listD, srcA[c], eArr, vals,
                                               thr, fbuf, zc);
  }

  // semantic attention mean scores, then beta-softmax + final linear (fused)
  fill_u32_kernel<<<1, 256, 0, stream>>>((unsigned*)wmean, 0u, 2);
  sem_kernel<<<(Nn * 2) / 8, 256, 0, stream>>>(z0, z1, semW1, semB1, semW2, wmean);
  final_kernel<<<Nn / 8, 256, 0, stream>>>(z0, z1, wmean, linW, linB, out);
}
